// QAttention_16647293239989
// MI455X (gfx1250) — matrix operand order repaired and checked by reading, not compiled, not run on hardware
//
#include <hip/hip_runtime.h>

typedef __attribute__((ext_vector_type(16))) __bf16 v16bf;
typedef __attribute__((ext_vector_type(8)))  __bf16 v8bf;
typedef __attribute__((ext_vector_type(8)))  float  v8f;
typedef __attribute__((ext_vector_type(4)))  int    v4i;

#define DIMC   768
#define NH     12
#define DH     64
#define INNER  768
#define BATCH  8
#define SEQ    1024
#define ATTN_SCALE 0.125f   // 64^-0.5
#define LDSB_PITCH 36       // 64 rows x 32 floats, padded to 36 to spread banks

// ---- CDNA5 async global->LDS staging (guarded; falls back to vector copy) ----
#ifdef __HIP_DEVICE_COMPILE__
#if __has_builtin(__builtin_amdgcn_global_load_async_to_lds_b128) && \
    __has_builtin(__builtin_amdgcn_s_wait_asynccnt)
#define USE_ASYNC_LDS 1
#endif
#endif
#ifndef USE_ASYNC_LDS
#define USE_ASYNC_LDS 0
#endif

static __device__ __forceinline__ v8f wmma_bf16(v16bf a, v16bf b, v8f c) {
  // D = A(16x32 bf16) * B(32x16 bf16) + C(16x16 f32)
  return __builtin_amdgcn_wmma_f32_16x16x32_bf16(false, a, false, b, (short)0, c,
                                                 false, false);
}

// Issue (no wait) a 64-row x 32-float fp32 tile copy (row stride DIMC) into
// LDS (pitch 36).  256 threads, 2 x 16B segments each = 8 KB.
static __device__ __forceinline__ void stage_issue(const float* __restrict__ wsrc,
                                                   float* __restrict__ lds,
                                                   int tid) {
  int seg = tid * 2;
#pragma unroll
  for (int u = 0; u < 2; ++u, ++seg) {
    const int row = seg >> 3;
    const int cs  = (seg & 7) * 4;
    const float* gp = wsrc + (size_t)row * DIMC + cs;
    float* lp = lds + row * LDSB_PITCH + cs;
#if USE_ASYNC_LDS
    __builtin_amdgcn_global_load_async_to_lds_b128(
        (v4i*)(void*)gp,
        (__attribute__((address_space(3))) v4i*)lp, 0, 0);
#else
    *(float4*)lp = *(const float4*)gp;
#endif
  }
}

static __device__ __forceinline__ void stage_wait() {
#if USE_ASYNC_LDS
  __builtin_amdgcn_s_wait_asynccnt(0);
#endif
}

// ---------------------------------------------------------------------------
// Kernel 1: QKV projection.  qkv = x(8192x768) @ w_qkv^T(768x2304), split into
// Q (scaled, bf16, [B,H,N,dh]), K (bf16, [B,H,N,dh]), V^T (bf16, [B,H,dh,N]).
// Wave = 64x64 output (4x4 WMMA tiles); block = 8 waves sharing a
// double-buffered async-staged 64x32 weight tile (DMA overlapped with WMMA).
// ---------------------------------------------------------------------------
__global__ __launch_bounds__(256)
void qkv_gemm_kernel(const float* __restrict__ x, const float* __restrict__ w,
                     __bf16* __restrict__ Qb, __bf16* __restrict__ Kb,
                     __bf16* __restrict__ Vt) {
  __shared__ float lsB[2][64 * LDSB_PITCH];
  const int tid  = threadIdx.x;
  const int lane = tid & 31;
  const int wave = tid >> 5;
  const int lmod = lane & 15;
  const int half = lane >> 4;
  const int m0 = (blockIdx.y * 8 + wave) * 64;   // [0,8192)
  const int n0 = blockIdx.x * 64;                // [0,2304)

  const float* ar[4];
#pragma unroll
  for (int mt = 0; mt < 4; ++mt)
    ar[mt] = x + (size_t)(m0 + mt * 16 + lmod) * DIMC;

  const float* wbase = w + (size_t)n0 * DIMC;

  // Prologue: stage k-step 0 and wait for it.
  stage_issue(wbase, lsB[0], tid);
  stage_wait();
  __syncthreads();

  v8f acc[4][4] = {};
  int cur = 0;
  for (int ks = 0; ks < DIMC; ks += 32) {
    // Kick off DMA for the next weight tile while we compute on this one.
    if (ks + 32 < DIMC) stage_issue(wbase + ks + 32, lsB[cur ^ 1], tid);

    // A fragments (4 m-tiles), fp32 -> bf16
    v16bf afr[4];
#pragma unroll
    for (int mt = 0; mt < 4; ++mt) {
      const float* p = ar[mt] + ks;
      if (ks + 32 < DIMC) __builtin_prefetch(p + 32, 0, 1);
      float4 t0 = ((const float4*)(p + half * 16))[0];
      float4 t1 = ((const float4*)(p + half * 16))[1];
      float4 t2 = ((const float4*)(p + half * 16))[2];
      float4 t3 = ((const float4*)(p + half * 16))[3];
      float af[16] = {t0.x,t0.y,t0.z,t0.w, t1.x,t1.y,t1.z,t1.w,
                      t2.x,t2.y,t2.z,t2.w, t3.x,t3.y,t3.z,t3.w};
#pragma unroll
      for (int i = 0; i < 16; ++i) afr[mt][i] = (__bf16)af[i];
    }

    // B fragments from LDS (4 n-tiles); each feeds 4 WMMAs
#pragma unroll
    for (int nt = 0; nt < 4; ++nt) {
      const float* lp = &lsB[cur][(nt * 16 + lmod) * LDSB_PITCH + half * 16];
      float4 u0 = ((const float4*)lp)[0], u1 = ((const float4*)lp)[1];
      float4 u2 = ((const float4*)lp)[2], u3 = ((const float4*)lp)[3];
      float bf_[16] = {u0.x,u0.y,u0.z,u0.w, u1.x,u1.y,u1.z,u1.w,
                       u2.x,u2.y,u2.z,u2.w, u3.x,u3.y,u3.z,u3.w};
      v16bf b;
#pragma unroll
      for (int i = 0; i < 16; ++i) b[i] = (__bf16)bf_[i];
#pragma unroll
      for (int mt = 0; mt < 4; ++mt)
        acc[mt][nt] = wmma_bf16(afr[mt], b, acc[mt][nt]);
    }

    // Retire the DMA for the next tile, then flip buffers.
    stage_wait();
    __syncthreads();
    cur ^= 1;
  }

  // Epilogue: n-supertile (64 wide, 64-aligned) is exactly one (s,h) pair;
  // the s-branch is uniform, hoisted out of the store loops.
  const int s = n0 / INNER;                 // 0=Q 1=K 2=V
  const int h = (n0 % INNER) / DH;
  const int bidx = m0 >> 10;                // 64-row tile stays in one batch
  const size_t bhI = (size_t)bidx * NH + h;
  const int nqBase = (m0 & 1023) + half * 8;
  if (s == 2) {
#pragma unroll
    for (int mt = 0; mt < 4; ++mt)
#pragma unroll
      for (int nt = 0; nt < 4; ++nt) {
        __bf16* vdst = Vt + (bhI * DH + nt * 16 + lmod) * SEQ + nqBase + mt * 16;
#pragma unroll
        for (int r = 0; r < 8; ++r) vdst[r] = (__bf16)acc[mt][nt][r];
      }
  } else {
    __bf16* base = (s == 0) ? Qb : Kb;
    const float scl = (s == 0) ? ATTN_SCALE : 1.0f;
#pragma unroll
    for (int mt = 0; mt < 4; ++mt)
#pragma unroll
      for (int nt = 0; nt < 4; ++nt) {
        const int d = nt * 16 + lmod;
#pragma unroll
        for (int r = 0; r < 8; ++r)
          base[(bhI * SEQ + nqBase + mt * 16 + r) * DH + d] =
              (__bf16)(acc[mt][nt][r] * scl);
      }
  }
}

// ---------------------------------------------------------------------------
// Kernel 2: flash-style attention.  One wave per (b,h, 32 query rows): K and V
// fragments amortized over 2 q-tiles.  Per 32-key chunk: 8 score WMMA, fp32
// online softmax, P re-striped via LDS, 8 PV WMMA.  Out: [B,N,H*dh] bf16.
// ---------------------------------------------------------------------------
__global__ __launch_bounds__(32)
void attn_kernel(const __bf16* __restrict__ Qb, const __bf16* __restrict__ Kb,
                 const __bf16* __restrict__ Vt, __bf16* __restrict__ AOb) {
  __shared__ __bf16 pT[16 * 32];     // P tile bounce: C-layout -> A-layout
  const int lane = threadIdx.x & 31;
  const int lmod = lane & 15;
  const int half = lane >> 4;
  const int bh = blockIdx.y;         // b*12 + h
  const int q0 = blockIdx.x * 32;    // first query row

  // Q fragments: 2 q-tiles x 2 k-steps over dh=64 (A-layout)
  v16bf aq[2][2];
#pragma unroll
  for (int qi = 0; qi < 2; ++qi) {
    const __bf16* qrow = Qb + ((size_t)bh * SEQ + q0 + qi * 16 + lmod) * DH;
#pragma unroll
    for (int j = 0; j < 2; ++j) {
      v8bf lo = *(const v8bf*)(qrow + j * 32 + half * 8);
      v8bf hi = *(const v8bf*)(qrow + j * 32 + 16 + half * 8);
#pragma unroll
      for (int i = 0; i < 8; ++i) { aq[qi][j][i] = lo[i]; aq[qi][j][8 + i] = hi[i]; }
    }
  }

  float mrun[2][8], lrun[2][8];
  v8f acc[2][4] = {};
#pragma unroll
  for (int qi = 0; qi < 2; ++qi)
#pragma unroll
    for (int r = 0; r < 8; ++r) { mrun[qi][r] = -1e30f; lrun[qi][r] = 0.0f; }

  for (int kc = 0; kc < SEQ / 32; ++kc) {
    // ---- S = Q @ K^T for 32 keys; each K fragment feeds both q-tiles ----
    v8f s[2][2] = {};
#pragma unroll
    for (int f = 0; f < 2; ++f) {
      const __bf16* krow = Kb + ((size_t)bh * SEQ + kc * 32 + f * 16 + lmod) * DH;
#pragma unroll
      for (int j = 0; j < 2; ++j) {
        v16bf bk;
        v8bf lo = *(const v8bf*)(krow + j * 32 + half * 8);
        v8bf hi = *(const v8bf*)(krow + j * 32 + 16 + half * 8);
#pragma unroll
        for (int i = 0; i < 8; ++i) { bk[i] = lo[i]; bk[8 + i] = hi[i]; }
#pragma unroll
        for (int qi = 0; qi < 2; ++qi)
          s[qi][f] = wmma_bf16(aq[qi][j], bk, s[qi][f]);
      }
    }

    // ---- V fragments for this chunk, held for both q-tiles ----
    v16bf bv[4];
#pragma unroll
    for (int t = 0; t < 4; ++t) {
      const __bf16* vrow =
          Vt + ((size_t)bh * DH + t * 16 + lmod) * SEQ + kc * 32 + half * 8;
      v8bf lo = *(const v8bf*)(vrow);
      v8bf hi = *(const v8bf*)(vrow + 16);
#pragma unroll
      for (int i = 0; i < 8; ++i) { bv[t][i] = lo[i]; bv[t][8 + i] = hi[i]; }
    }

#pragma unroll
    for (int qi = 0; qi < 2; ++qi) {
      // ---- fp32 online softmax (rows lane-local: m = half*8 + r) ----
      float p0[8], p1[8];
#pragma unroll
      for (int r = 0; r < 8; ++r) {
        float cm = fmaxf(s[qi][0][r], s[qi][1][r]);
        cm = fmaxf(cm, __shfl_xor(cm, 1, 32));
        cm = fmaxf(cm, __shfl_xor(cm, 2, 32));
        cm = fmaxf(cm, __shfl_xor(cm, 4, 32));
        cm = fmaxf(cm, __shfl_xor(cm, 8, 32));
        const float mnew = fmaxf(mrun[qi][r], cm);
        const float sc = __expf(mrun[qi][r] - mnew);
        lrun[qi][r] *= sc;
        acc[qi][0][r] *= sc; acc[qi][1][r] *= sc;
        acc[qi][2][r] *= sc; acc[qi][3][r] *= sc;
        p0[r] = __expf(s[qi][0][r] - mnew);
        p1[r] = __expf(s[qi][1][r] - mnew);
        float rs = p0[r] + p1[r];
        rs += __shfl_xor(rs, 1, 32);
        rs += __shfl_xor(rs, 2, 32);
        rs += __shfl_xor(rs, 4, 32);
        rs += __shfl_xor(rs, 8, 32);
        lrun[qi][r] += rs;
        mrun[qi][r] = mnew;
      }

      // ---- re-stripe P: C-layout -> A-layout via LDS (single wave) ----
      __syncthreads();
#pragma unroll
      for (int r = 0; r < 8; ++r) {
        const int m = half * 8 + r;
        pT[m * 32 + lmod]      = (__bf16)p0[r];
        pT[m * 32 + 16 + lmod] = (__bf16)p1[r];
      }
      __syncthreads();
      v16bf ap;
#pragma unroll
      for (int i = 0; i < 16; ++i) {
        const int k = (i >> 3) * 16 + half * 8 + (i & 7);
        ap[i] = pT[lmod * 32 + k];
      }

      // ---- O += P @ V ----
#pragma unroll
      for (int t = 0; t < 4; ++t)
        acc[qi][t] = wmma_bf16(ap, bv[t], acc[qi][t]);
    }
  }

  // ---- normalize and store attn-out as bf16 [B,N,H*dh] ----
  const int b = bh / NH, h = bh % NH;
#pragma unroll
  for (int qi = 0; qi < 2; ++qi) {
    float rl[8];
#pragma unroll
    for (int r = 0; r < 8; ++r) rl[r] = 1.0f / lrun[qi][r];
#pragma unroll
    for (int t = 0; t < 4; ++t) {
#pragma unroll
      for (int r = 0; r < 8; ++r) {
        const int m = q0 + qi * 16 + half * 8 + r;
        AOb[((size_t)b * SEQ + m) * INNER + h * DH + t * 16 + lmod] =
            (__bf16)(acc[qi][t][r] * rl[r]);
      }
    }
  }
}

// ---------------------------------------------------------------------------
// Kernel 3: output projection. out(8192x768 f32) = AO(bf16) @ w_proj^T.
// Same 64x64-per-wave blocking; double-buffered async-staged weight tile.
// ---------------------------------------------------------------------------
__global__ __launch_bounds__(256)
void proj_gemm_kernel(const __bf16* __restrict__ AOb,
                      const float* __restrict__ wproj, float* __restrict__ out) {
  __shared__ float lsB[2][64 * LDSB_PITCH];
  const int tid  = threadIdx.x;
  const int lane = tid & 31;
  const int wave = tid >> 5;
  const int lmod = lane & 15;
  const int half = lane >> 4;
  const int m0 = (blockIdx.y * 8 + wave) * 64;   // [0,8192)
  const int n0 = blockIdx.x * 64;                // [0,768)

  const __bf16* ar[4];
#pragma unroll
  for (int mt = 0; mt < 4; ++mt)
    ar[mt] = AOb + (size_t)(m0 + mt * 16 + lmod) * INNER;

  const float* wbase = wproj + (size_t)n0 * DIMC;

  stage_issue(wbase, lsB[0], tid);
  stage_wait();
  __syncthreads();

  v8f acc[4][4] = {};
  int cur = 0;
  for (int ks = 0; ks < INNER; ks += 32) {
    if (ks + 32 < INNER) stage_issue(wbase + ks + 32, lsB[cur ^ 1], tid);

    v16bf afr[4];
#pragma unroll
    for (int mt = 0; mt < 4; ++mt) {
      const __bf16* p = ar[mt] + ks;
      if (ks + 32 < INNER) __builtin_prefetch(p + 32, 0, 1);
      v8bf lo = *(const v8bf*)(p + half * 16);
      v8bf hi = *(const v8bf*)(p + half * 16 + 8);
#pragma unroll
      for (int i = 0; i < 8; ++i) { afr[mt][i] = lo[i]; afr[mt][8 + i] = hi[i]; }
    }
#pragma unroll
    for (int nt = 0; nt < 4; ++nt) {
      const float* lp = &lsB[cur][(nt * 16 + lmod) * LDSB_PITCH + half * 16];
      float4 u0 = ((const float4*)lp)[0], u1 = ((const float4*)lp)[1];
      float4 u2 = ((const float4*)lp)[2], u3 = ((const float4*)lp)[3];
      float bf_[16] = {u0.x,u0.y,u0.z,u0.w, u1.x,u1.y,u1.z,u1.w,
                       u2.x,u2.y,u2.z,u2.w, u3.x,u3.y,u3.z,u3.w};
      v16bf b;
#pragma unroll
      for (int i = 0; i < 16; ++i) b[i] = (__bf16)bf_[i];
#pragma unroll
      for (int mt = 0; mt < 4; ++mt)
        acc[mt][nt] = wmma_bf16(afr[mt], b, acc[mt][nt]);
    }

    stage_wait();
    __syncthreads();
    cur ^= 1;
  }
#pragma unroll
  for (int mt = 0; mt < 4; ++mt)
#pragma unroll
    for (int nt = 0; nt < 4; ++nt)
#pragma unroll
      for (int r = 0; r < 8; ++r)
        out[(size_t)(m0 + mt * 16 + half * 8 + r) * DIMC + n0 + nt * 16 + lmod] =
            acc[mt][nt][r];
}

// ---------------------------------------------------------------------------
extern "C" void kernel_launch(void* const* d_in, const int* in_sizes, int n_in,
                              void* d_out, int out_size, void* d_ws, size_t ws_size,
                              hipStream_t stream) {
  const float* x      = (const float*)d_in[0];
  const float* w_qkv  = (const float*)d_in[1];
  const float* w_proj = (const float*)d_in[2];
  float* out = (float*)d_out;

  char* ws = (char*)d_ws;
  const size_t TBYTES = (size_t)BATCH * NH * SEQ * DH * sizeof(__bf16); // 12.6 MB
  __bf16* Qb  = (__bf16*)(ws + 0 * TBYTES);
  __bf16* Kb  = (__bf16*)(ws + 1 * TBYTES);
  __bf16* Vt  = (__bf16*)(ws + 2 * TBYTES);
  __bf16* AOb = (__bf16*)(ws + 3 * TBYTES);

  // 1) QKV: 36 n-supertiles x (128 m-supertiles / 8 waves)
  dim3 gA(36, 16);
  qkv_gemm_kernel<<<gA, 256, 0, stream>>>(x, w_qkv, Qb, Kb, Vt);

  // 2) attention: 32 query supertiles x 96 (b,h) pairs, 1 wave/block
  dim3 gB(32, BATCH * NH);
  attn_kernel<<<gB, 32, 0, stream>>>(Qb, Kb, Vt, AOb);

  // 3) projection: 12 n-supertiles x 16 blocks
  dim3 gC(12, 16);
  proj_gemm_kernel<<<gC, 256, 0, stream>>>(AOb, w_proj, out);
}
